// PathPredictor_74397423501961
// MI455X (gfx1250) — matrix operand order repaired and checked by reading, not compiled, not run on hardware
//
#include <hip/hip_runtime.h>
#include <cmath>

typedef __attribute__((ext_vector_type(16))) __bf16 v16bf;
typedef __attribute__((ext_vector_type(8)))  __bf16 v8bf;
typedef __attribute__((ext_vector_type(8)))  float  v8f;

namespace {
constexpr int Hc     = 256;     // hidden
constexpr int NRELc  = 460;     // relations
constexpr int RELP16 = 464;     // NREL padded to 16 rows (M padding for mapped_rel)
constexpr int RELPK  = 480;     // NREL padded to 32 (K-dim padding for attn)
constexpr int RELN   = 512;     // NREL padded to 64 (N-dim padding for attn_base)
constexpr int NBc    = 4096;    // batch
constexpr int NENTc  = 20000;   // entities (= 250 * 80)
constexpr int CURTSc = 10;
constexpr int SP_PARTS   = (NBc / 16) * (NENTc / 80);   // 256*250 = 64000 waves
constexpr int MATCH_BLKS = 1024;
}

// ---------------- device helpers ----------------

__device__ __forceinline__ float wave_sum(float v) {
  #pragma unroll
  for (int off = 16; off > 0; off >>= 1) v += __shfl_down(v, off, 32);
  return v;
}

union AFrag { v16bf v; v8bf h[2]; };

__device__ __forceinline__ v8f wmma_bf16(v16bf a, v16bf b, v8f c) {
  return __builtin_amdgcn_wmma_f32_16x16x32_bf16(false, a, false, b, (short)0, c, false, false);
}

// ---------------- generic WMMA GEMM:  C[M,N] = A[M,K](bf16) @ W[N,K](bf16)^T (+ bias) ----------
// wave tile: 16 rows x 64 cols (4 subtiles). REQUIRES N % 64 == 0, M % 16 == 0, K % 32 == 0.
// Tile id scalarized via readfirstlane -> all guards & address bases in SGPRs, no exec masking.
template <typename OutT, int K, bool BIAS>
__global__ __launch_bounds__(256)
void gemm_wmma(const __bf16* __restrict__ A, int lda,
               const __bf16* __restrict__ W, int ldw,
               const float* __restrict__ bias,
               OutT* __restrict__ C, int ldc,
               int M, int N)
{
  const int lane = threadIdx.x & 31;
  const int wid  = __builtin_amdgcn_readfirstlane(blockIdx.x * 8 + (threadIdx.x >> 5));
  const int tilesN = N >> 6;
  const int tm = wid / tilesN;
  if (tm >= (M >> 4)) return;                     // scalar branch
  const int m0 = tm << 4;
  const int n0 = (wid % tilesN) << 6;
  const int hi16 = lane >> 4, lane15 = lane & 15;

  v8f acc[4] = {{}, {}, {}, {}};
  const __bf16* arow = A + (size_t)(m0 + lane15) * lda + (hi16 << 4);
  const __bf16* brow = W + (size_t)(n0 + lane15) * ldw + (hi16 << 4);

  #pragma unroll
  for (int kk = 0; kk < K; kk += 32) {
    AFrag af;
    af.h[0] = *(const v8bf*)(arow + kk);          // K = kk + hi*16 + [0..7]
    af.h[1] = *(const v8bf*)(arow + kk + 8);      // K = kk + hi*16 + 8 + [0..7]
    #pragma unroll
    for (int j = 0; j < 4; ++j) {
      v16bf b = *(const v16bf*)(brow + (size_t)(j << 4) * ldw + kk);
      acc[j] = wmma_bf16(af.v, b, acc[j]);
    }
  }

  const int rbase = m0 + (hi16 << 3);             // C: vgpr r -> row rbase+r, col = lane15
  #pragma unroll
  for (int j = 0; j < 4; ++j) {
    const int col = n0 + (j << 4) + lane15;
    const float bv = BIAS ? bias[col] : 0.0f;
    OutT* cp = C + (size_t)rbase * ldc + col;
    #pragma unroll
    for (int r = 0; r < 8; ++r) cp[(size_t)r * ldc] = (OutT)(acc[j][r] + bv);
  }
}

// ---------------- score GEMM: 16x80 wave tile (20000 = 250*80, guard-free) ----------------
// out = sigmoid(sh2 @ aligned^T), fused stable-softplus per-wave partial sums.
__global__ __launch_bounds__(256)
void score_wmma(const __bf16* __restrict__ A,      // sh2 bf16 [4096 x 256]
                const __bf16* __restrict__ W,      // aligned bf16 [20000 x 256]
                float* __restrict__ out,           // sigmoid(score) [4096 x 20000]
                float* __restrict__ sp_part)       // per-wave softplus partials [SP_PARTS]
{
  constexpr int K = Hc, N = NENTc;
  const int lane = threadIdx.x & 31;
  const int wid  = __builtin_amdgcn_readfirstlane(blockIdx.x * 8 + (threadIdx.x >> 5));
  constexpr int tilesN = N / 80;                   // 250
  const int tm = wid / tilesN;
  if (tm >= (NBc >> 4)) return;
  const int m0 = tm << 4;
  const int n0 = (wid % tilesN) * 80;
  const int hi16 = lane >> 4, lane15 = lane & 15;

  v8f acc[5] = {{}, {}, {}, {}, {}};
  const __bf16* arow = A + (size_t)(m0 + lane15) * K + (hi16 << 4);
  const __bf16* brow = W + (size_t)(n0 + lane15) * K + (hi16 << 4);

  #pragma unroll
  for (int kk = 0; kk < K; kk += 32) {
    AFrag af;
    af.h[0] = *(const v8bf*)(arow + kk);
    af.h[1] = *(const v8bf*)(arow + kk + 8);
    #pragma unroll
    for (int j = 0; j < 5; ++j) {
      v16bf b = *(const v16bf*)(brow + (size_t)(j << 4) * K + kk);
      acc[j] = wmma_bf16(af.v, b, acc[j]);
    }
  }

  const int rbase = m0 + (hi16 << 3);
  float sp = 0.0f;
  #pragma unroll
  for (int j = 0; j < 5; ++j) {
    const int col = n0 + (j << 4) + lane15;
    float* cp = out + (size_t)rbase * N + col;
    #pragma unroll
    for (int r = 0; r < 8; ++r) {
      float x = acc[j][r];
      cp[(size_t)r * N] = 1.0f / (1.0f + expf(-x));            // sigmoid
      sp += fmaxf(x, 0.0f) + log1pf(expf(-fabsf(x)));          // stable softplus
    }
  }
  sp = wave_sum(sp);
  if (lane == 0) sp_part[wid] = sp;
}

// ---------------- elementwise / utility kernels ----------------

__global__ void cvt_bf16(const float* __restrict__ s, __bf16* __restrict__ d, int n) {
  int i = blockIdx.x * blockDim.x + threadIdx.x;
  if (i < n) d[i] = (__bf16)s[i];
}

__global__ void cvt_bf16_padrows(const float* __restrict__ s, __bf16* __restrict__ d,
                                 int rows, int cols, int prows) {
  int i = blockIdx.x * blockDim.x + threadIdx.x;
  if (i >= prows * cols) return;
  int r = i / cols;
  d[i] = (r < rows) ? (__bf16)s[i] : (__bf16)0.0f;
}

// mapped_rel[460(+pad)x256] (f32) -> mappedT[256 x 480] (bf16, K-padded with zeros)
__global__ void transpose_pad(const float* __restrict__ m, __bf16* __restrict__ mT) {
  int i = blockIdx.x * blockDim.x + threadIdx.x;
  if (i >= Hc * RELPK) return;
  int n = i / RELPK, k = i % RELPK;
  mT[i] = (k < NRELc) ? (__bf16)m[(size_t)k * Hc + n] : (__bf16)0.0f;
}

__global__ void gather_rows(const float* __restrict__ src, const int* __restrict__ tri, int sel,
                            float* __restrict__ df, __bf16* __restrict__ db, int n) {
  int i = blockIdx.x * blockDim.x + threadIdx.x;
  if (i >= n) return;
  int row = i >> 8, c = i & 255;
  float v = src[(size_t)tri[row * 3 + sel] * Hc + c];
  df[i] = v;
  if (db) db[i] = (__bf16)v;
}

__global__ void zero_h(float* __restrict__ hf, __bf16* __restrict__ hb, int n) {
  int i = blockIdx.x * blockDim.x + threadIdx.x;
  if (i < n) { hf[i] = 0.0f; hb[i] = (__bf16)0.0f; }
}

// masked softmax over 460 rel-logits per row -> bf16 attn (padded to 480 with zeros)
__global__ __launch_bounds__(256)
void softmax_attn(const float* __restrict__ part, const float* __restrict__ base,
                  __bf16* __restrict__ attn, int t) {
  __shared__ float red[256];
  const int row = blockIdx.x, tid = threadIdx.x;
  const float* pr = part + (size_t)row * (CURTSc * NRELc) + t * NRELc;
  const float* br = base + (size_t)row * RELN;
  const int k0 = tid, k1 = tid + 256;
  float v0 = -1e30f, v1 = -1e30f;
  if (k0 < NRELc) { v0 = pr[k0] * br[k0]; if (v0 == 0.0f) v0 = -1e9f; }
  if (k1 < NRELc) { v1 = pr[k1] * br[k1]; if (v1 == 0.0f) v1 = -1e9f; }
  red[tid] = fmaxf(v0, v1); __syncthreads();
  #pragma unroll
  for (int s = 128; s > 0; s >>= 1) { if (tid < s) red[tid] = fmaxf(red[tid], red[tid + s]); __syncthreads(); }
  const float mx = red[0]; __syncthreads();
  const float e0 = (k0 < NRELc) ? expf(v0 - mx) : 0.0f;
  const float e1 = (k1 < NRELc) ? expf(v1 - mx) : 0.0f;
  red[tid] = e0 + e1; __syncthreads();
  #pragma unroll
  for (int s = 128; s > 0; s >>= 1) { if (tid < s) red[tid] += red[tid + s]; __syncthreads(); }
  const float inv = 1.0f / red[0];
  __bf16* ar = attn + (size_t)row * RELPK;
  if (k0 < NRELc) ar[k0] = (__bf16)(e0 * inv);
  if (k1 < RELPK) ar[k1] = (k1 < NRELc) ? (__bf16)(e1 * inv) : (__bf16)0.0f;
}

// GRU gates: gi/gh are [4096 x 768] with (r,z,n) blocks of 256
__global__ void gru_gate(const float* __restrict__ gi, const float* __restrict__ gh,
                         const float* __restrict__ hprev,
                         float* __restrict__ hf, __bf16* __restrict__ hb, int n) {
  int i = blockIdx.x * blockDim.x + threadIdx.x;
  if (i >= n) return;
  int row = i >> 8, c = i & 255;
  const float* gir = gi + (size_t)row * 768;
  const float* ghr = gh + (size_t)row * 768;
  float r  = 1.0f / (1.0f + expf(-(gir[c]       + ghr[c])));
  float z  = 1.0f / (1.0f + expf(-(gir[256 + c] + ghr[256 + c])));
  float nn = tanhf(gir[512 + c] + r * ghr[512 + c]);
  float h  = (1.0f - z) * nn + z * hprev[i];
  hf[i] = h; hb[i] = (__bf16)h;
}

__global__ void ph_combine(const float* __restrict__ ph2, const float* __restrict__ qrel,
                           float* __restrict__ pf, __bf16* __restrict__ pb, int n) {
  int i = blockIdx.x * blockDim.x + threadIdx.x;
  if (i >= n) return;
  float v = 0.1f * ph2[i] + qrel[i];
  pf[i] = v; pb[i] = (__bf16)v;
}

__global__ void mul_elem(const float* __restrict__ a, const float* __restrict__ b,
                         float* __restrict__ cf, __bf16* __restrict__ cb, int n) {
  int i = blockIdx.x * blockDim.x + threadIdx.x;
  if (i >= n) return;
  float v = a[i] * b[i];
  cf[i] = v; cb[i] = (__bf16)v;
}

// per-block partial of sum((a-b)^2)
__global__ __launch_bounds__(256)
void match_partial(const float* __restrict__ a, const float* __restrict__ b,
                   float* __restrict__ part, int n) {
  __shared__ float red[256];
  float s = 0.0f;
  for (int i = blockIdx.x * blockDim.x + threadIdx.x; i < n; i += gridDim.x * blockDim.x) {
    float d = a[i] - b[i]; s += d * d;
  }
  red[threadIdx.x] = s; __syncthreads();
  #pragma unroll
  for (int k = 128; k > 0; k >>= 1) { if (threadIdx.x < k) red[threadIdx.x] += red[threadIdx.x + k]; __syncthreads(); }
  if (threadIdx.x == 0) part[blockIdx.x] = red[0];
}

// exact f32 pos_logits: one wave per row
__global__ __launch_bounds__(256)
void pos_partial(const float* __restrict__ sh2, const float* __restrict__ aligned,
                 const int* __restrict__ tri, float* __restrict__ pos_part) {
  const int lane = threadIdx.x & 31;
  const int row  = __builtin_amdgcn_readfirstlane(blockIdx.x * 8 + (threadIdx.x >> 5));
  if (row >= NBc) return;
  const int obj = tri[row * 3 + 2];
  const float* x = sh2 + (size_t)row * Hc;
  const float* y = aligned + (size_t)obj * Hc;
  float s = 0.0f;
  #pragma unroll
  for (int c = lane; c < Hc; c += 32) s += x[c] * y[c];
  s = wave_sum(s);
  if (lane == 0) pos_part[row] = s;
}

// deterministic final reduction -> d_out[0], d_out[1]
__global__ __launch_bounds__(256)
void finalize(const float* __restrict__ match_part, const float* __restrict__ sp_part,
              const float* __restrict__ pos_part, float* __restrict__ out) {
  __shared__ double red[256];
  const int tid = threadIdx.x;
  double s = 0.0;
  for (int i = tid; i < MATCH_BLKS; i += 256) s += (double)match_part[i];
  red[tid] = s; __syncthreads();
  #pragma unroll
  for (int k = 128; k > 0; k >>= 1) { if (tid < k) red[tid] += red[tid + k]; __syncthreads(); }
  const double sq = red[0]; __syncthreads();

  s = 0.0;
  for (int i = tid; i < SP_PARTS; i += 256) s += (double)sp_part[i];
  red[tid] = s; __syncthreads();
  #pragma unroll
  for (int k = 128; k > 0; k >>= 1) { if (tid < k) red[tid] += red[tid + k]; __syncthreads(); }
  const double sp = red[0]; __syncthreads();

  s = 0.0;
  for (int i = tid; i < NBc; i += 256) s += (double)pos_part[i];
  red[tid] = s; __syncthreads();
  #pragma unroll
  for (int k = 128; k > 0; k >>= 1) { if (tid < k) red[tid] += red[tid + k]; __syncthreads(); }
  const double pos = red[0];

  if (tid == 0) {
    out[0] = (float)(sq / ((double)NBc * Hc));
    out[1] = (float)((sp - pos) / ((double)NBc * NENTc));
  }
}

// ---------------- host ----------------

static inline int cdiv(int a, int b) { return (a + b - 1) / b; }
static inline int gemm_blocks(int M, int N) {        // N % 64 == 0
  return cdiv((M >> 4) * (N >> 6), 8);
}

extern "C" void kernel_launch(void* const* d_in, const int* in_sizes, int n_in,
                              void* d_out, int out_size, void* d_ws, size_t ws_size,
                              hipStream_t stream) {
  (void)in_sizes; (void)n_in; (void)out_size; (void)ws_size;

  const float* pre_emb = (const float*)d_in[0];
  const float* r_emb   = (const float*)d_in[1];
  const float* part    = (const float*)d_in[2];
  const float* W_map1  = (const float*)d_in[3];
  const float* b_map1  = (const float*)d_in[4];
  const float* W_map2  = (const float*)d_in[5];
  const float* b_map2  = (const float*)d_in[6];
  const float* W_h1    = (const float*)d_in[7];
  const float* b_h1    = (const float*)d_in[8];
  const float* W_h2    = (const float*)d_in[9];
  const float* b_h2    = (const float*)d_in[10];
  const float* W_attn  = (const float*)d_in[11];
  const float* b_attn  = (const float*)d_in[12];
  const float* W_align = (const float*)d_in[13];
  const float* b_align = (const float*)d_in[14];
  const float* W_ih    = (const float*)d_in[15];
  const float* W_hh    = (const float*)d_in[16];
  const float* b_ih    = (const float*)d_in[17];
  const float* b_hh    = (const float*)d_in[18];
  const int*   tri     = (const int*)d_in[19];
  float* out = (float*)d_out;      // [match_loss, path_loss, score(4096x20000)]
  float* score = out + 2;

  // ---- bump allocator over workspace (same layout every call: deterministic) ----
  char* w = (char*)d_ws;
  auto alloc = [&](size_t bytes) -> void* {
    void* p = (void*)w; w += (bytes + 255) & ~(size_t)255; return p;
  };
  __bf16* W_map1_b  = (__bf16*)alloc((size_t)512 * 256 * 2);
  __bf16* W_map2_b  = (__bf16*)alloc((size_t)256 * 512 * 2);
  __bf16* W_h1_b    = (__bf16*)alloc((size_t)256 * 256 * 2);
  __bf16* W_h2_b    = (__bf16*)alloc((size_t)256 * 256 * 2);
  __bf16* W_attn_b  = (__bf16*)alloc((size_t)256 * 256 * 2);
  __bf16* W_align_b = (__bf16*)alloc((size_t)256 * 256 * 2);
  __bf16* W_ih_b    = (__bf16*)alloc((size_t)768 * 256 * 2);
  __bf16* W_hh_b    = (__bf16*)alloc((size_t)768 * 256 * 2);
  __bf16* r_emb_b   = (__bf16*)alloc((size_t)RELP16 * 256 * 2);
  __bf16* pre_b     = (__bf16*)alloc((size_t)NENTc * 256 * 2);
  __bf16* t1_b      = (__bf16*)alloc((size_t)RELP16 * 512 * 2);
  float*  mapped_f  = (float*)alloc((size_t)RELP16 * 256 * 4);
  __bf16* mapped_b  = (__bf16*)alloc((size_t)RELN * 256 * 2);    // 512 rows, >=460 zero
  __bf16* mappedT_b = (__bf16*)alloc((size_t)256 * RELPK * 2);
  float*  qrel_f    = (float*)alloc((size_t)NBc * 256 * 4);
  __bf16* qrel_b    = (__bf16*)alloc((size_t)NBc * 256 * 2);
  __bf16* qa_b      = (__bf16*)alloc((size_t)NBc * 256 * 2);
  float*  abase_f   = (float*)alloc((size_t)NBc * RELN * 4);     // ld 512
  __bf16* attn_b    = (__bf16*)alloc((size_t)NBc * RELPK * 2);
  __bf16* x_b       = (__bf16*)alloc((size_t)NBc * 256 * 2);
  float*  gi_f      = (float*)alloc((size_t)NBc * 768 * 4);
  float*  gh_f      = (float*)alloc((size_t)NBc * 768 * 4);
  float*  h_f       = (float*)alloc((size_t)NBc * 256 * 4);
  __bf16* h_b       = (__bf16*)alloc((size_t)NBc * 256 * 2);
  __bf16* ph1_b     = (__bf16*)alloc((size_t)NBc * 256 * 2);
  float*  ph2_f     = (float*)alloc((size_t)NBc * 256 * 4);
  float*  ph_f      = (float*)alloc((size_t)NBc * 256 * 4);
  __bf16* ph_b      = (__bf16*)alloc((size_t)NBc * 256 * 2);
  float*  h2_f      = (float*)alloc((size_t)NBc * 256 * 4);
  __bf16* h2_b      = (__bf16*)alloc((size_t)NBc * 256 * 2);
  float*  aligned_f = (float*)alloc((size_t)NENTc * 256 * 4);
  __bf16* aligned_b = (__bf16*)alloc((size_t)NENTc * 256 * 2);
  float*  sub_f     = (float*)alloc((size_t)NBc * 256 * 4);
  float*  sh2_f     = (float*)alloc((size_t)NBc * 256 * 4);
  __bf16* sh2_b     = (__bf16*)alloc((size_t)NBc * 256 * 2);
  float*  mpart     = (float*)alloc((size_t)MATCH_BLKS * 4);
  float*  sppart    = (float*)alloc((size_t)SP_PARTS * 4);
  float*  pospart   = (float*)alloc((size_t)NBc * 4);

  const int NBH = NBc * 256;

  // ---- weight / embedding conversions to bf16 ----
  cvt_bf16<<<cdiv(512 * 256, 256), 256, 0, stream>>>(W_map1, W_map1_b, 512 * 256);
  cvt_bf16<<<cdiv(256 * 512, 256), 256, 0, stream>>>(W_map2, W_map2_b, 256 * 512);
  cvt_bf16<<<cdiv(256 * 256, 256), 256, 0, stream>>>(W_h1, W_h1_b, 256 * 256);
  cvt_bf16<<<cdiv(256 * 256, 256), 256, 0, stream>>>(W_h2, W_h2_b, 256 * 256);
  cvt_bf16<<<cdiv(256 * 256, 256), 256, 0, stream>>>(W_attn, W_attn_b, 256 * 256);
  cvt_bf16<<<cdiv(256 * 256, 256), 256, 0, stream>>>(W_align, W_align_b, 256 * 256);
  cvt_bf16<<<cdiv(768 * 256, 256), 256, 0, stream>>>(W_ih, W_ih_b, 768 * 256);
  cvt_bf16<<<cdiv(768 * 256, 256), 256, 0, stream>>>(W_hh, W_hh_b, 768 * 256);
  cvt_bf16<<<cdiv(NENTc * 256, 256), 256, 0, stream>>>(pre_emb, pre_b, NENTc * 256);
  cvt_bf16_padrows<<<cdiv(RELP16 * 256, 256), 256, 0, stream>>>(r_emb, r_emb_b, NRELc, 256, RELP16);

  // ---- mapped_rel = (r_emb @ W_map1^T + b1) @ W_map2^T + b2 ----
  gemm_wmma<__bf16, 256, true><<<gemm_blocks(RELP16, 512), 256, 0, stream>>>(
      r_emb_b, 256, W_map1_b, 256, b_map1, t1_b, 512, RELP16, 512);
  gemm_wmma<float, 512, true><<<gemm_blocks(RELP16, 256), 256, 0, stream>>>(
      t1_b, 512, W_map2_b, 512, b_map2, mapped_f, 256, RELP16, 256);
  // mapped_b: 512 rows, rows >= 460 zeroed (N-padding for attn_base GEMM)
  cvt_bf16_padrows<<<cdiv(RELN * 256, 256), 256, 0, stream>>>(mapped_f, mapped_b, NRELc, 256, RELN);
  transpose_pad<<<cdiv(256 * RELPK, 256), 256, 0, stream>>>(mapped_f, mappedT_b);

  // ---- q_rel gather; attn_base = (q_rel @ W_attn^T + b) @ mapped_rel^T ----
  gather_rows<<<cdiv(NBH, 256), 256, 0, stream>>>(mapped_f, tri, 1, qrel_f, qrel_b, NBH);
  gemm_wmma<__bf16, 256, true><<<gemm_blocks(NBc, 256), 256, 0, stream>>>(
      qrel_b, 256, W_attn_b, 256, b_attn, qa_b, 256, NBc, 256);
  gemm_wmma<float, 256, false><<<gemm_blocks(NBc, RELN), 256, 0, stream>>>(
      qa_b, 256, mapped_b, 256, nullptr, abase_f, RELN, NBc, RELN);

  // ---- GRU scan: t = 2..9 ----
  zero_h<<<cdiv(NBH, 256), 256, 0, stream>>>(h_f, h_b, NBH);
  for (int t = CURTSc - 8; t < CURTSc; ++t) {
    softmax_attn<<<NBc, 256, 0, stream>>>(part, abase_f, attn_b, t);
    gemm_wmma<__bf16, RELPK, false><<<gemm_blocks(NBc, 256), 256, 0, stream>>>(
        attn_b, RELPK, mappedT_b, RELPK, nullptr, x_b, 256, NBc, 256);
    gemm_wmma<float, 256, true><<<gemm_blocks(NBc, 768), 256, 0, stream>>>(
        x_b, 256, W_ih_b, 256, b_ih, gi_f, 768, NBc, 768);
    gemm_wmma<float, 256, true><<<gemm_blocks(NBc, 768), 256, 0, stream>>>(
        h_b, 256, W_hh_b, 256, b_hh, gh_f, 768, NBc, 768);
    gru_gate<<<cdiv(NBH, 256), 256, 0, stream>>>(gi_f, gh_f, h_f, h_f, h_b, NBH);
  }

  // ---- predicted_hist = 0.1*((q_rel@W_h1^T+b)@W_h2^T+b) + q_rel ----
  gemm_wmma<__bf16, 256, true><<<gemm_blocks(NBc, 256), 256, 0, stream>>>(
      qrel_b, 256, W_h1_b, 256, b_h1, ph1_b, 256, NBc, 256);
  gemm_wmma<float, 256, true><<<gemm_blocks(NBc, 256), 256, 0, stream>>>(
      ph1_b, 256, W_h2_b, 256, b_h2, ph2_f, 256, NBc, 256);
  ph_combine<<<cdiv(NBH, 256), 256, 0, stream>>>(ph2_f, qrel_f, ph_f, ph_b, NBH);

  // ---- match loss partials ----
  match_partial<<<MATCH_BLKS, 256, 0, stream>>>(ph_f, h_f, mpart, NBH);

  // ---- h2 = gru_step(q_rel, predicted_hist) ----
  gemm_wmma<float, 256, true><<<gemm_blocks(NBc, 768), 256, 0, stream>>>(
      qrel_b, 256, W_ih_b, 256, b_ih, gi_f, 768, NBc, 768);
  gemm_wmma<float, 256, true><<<gemm_blocks(NBc, 768), 256, 0, stream>>>(
      ph_b, 256, W_hh_b, 256, b_hh, gh_f, 768, NBc, 768);
  gru_gate<<<cdiv(NBH, 256), 256, 0, stream>>>(gi_f, gh_f, ph_f, h2_f, h2_b, NBH);

  // ---- aligned_all, sub, sh2 = sub*h2 ----
  gemm_wmma<float, 256, true><<<gemm_blocks(NENTc, 256), 256, 0, stream>>>(
      pre_b, 256, W_align_b, 256, b_align, aligned_f, 256, NENTc, 256);
  cvt_bf16<<<cdiv(NENTc * 256, 256), 256, 0, stream>>>(aligned_f, aligned_b, NENTc * 256);
  gather_rows<<<cdiv(NBH, 256), 256, 0, stream>>>(aligned_f, tri, 0, sub_f, nullptr, NBH);
  mul_elem<<<cdiv(NBH, 256), 256, 0, stream>>>(sub_f, h2_f, sh2_f, sh2_b, NBH);

  // ---- score = sigmoid(sh2 @ aligned^T), fused softplus partials (16x80 tiles) ----
  score_wmma<<<cdiv(SP_PARTS, 8), 256, 0, stream>>>(sh2_b, aligned_b, score, sppart);

  // ---- pos logits (exact f32), final losses ----
  pos_partial<<<NBc / 8, 256, 0, stream>>>(sh2_f, aligned_f, tri, pospart);
  finalize<<<1, 256, 0, stream>>>(mpart, sppart, pospart, out);
}
